// TransformerAgg_34737695490530
// MI455X (gfx1250) — matrix operand order repaired and checked by reading, not compiled, not run on hardware
//
#include <hip/hip_runtime.h>
#include <hip/hip_bf16.h>

// ---------------------------------------------------------------------------
// Problem constants (from reference)
// ---------------------------------------------------------------------------
#define NB   8192      // groups
#define NL   64        // padded rows per group
#define ND   66        // d_model
#define NE   262144    // edges
#define QLD  208       // 3*D=198 padded to 13 WMMA N-tiles
#define KPAD 96        // D=66 padded to 3 K-steps of 32
#define F1N  288       // 4*D=264 padded
#define ON   80        // D=66 padded to 5 N-tiles

typedef __attribute__((ext_vector_type(16))) __bf16 v16bf;
typedef __attribute__((ext_vector_type(8)))  float  v8f;

__device__ __forceinline__ __bf16 f2bf(float f) { return (__bf16)f; }

#define WMMA_BF16(a, b, c) \
  __builtin_amdgcn_wmma_f32_16x16x32_bf16(false, (a), false, (b), (short)0, (c), false, false)

// A fragment: 16x32 bf16 tile, row-major source with leading dim lda.
// lane: M = m0 + (lane&15); K base offset = ((lane>>4)<<3); element e -> K += (e&7)+((e>>3)<<4)
__device__ __forceinline__ v16bf load_a(const __bf16* A, int lda, int m0, int k0, int lane) {
  v16bf a;
  const __bf16* p = A + (m0 + (lane & 15)) * lda + k0 + ((lane >> 4) << 3);
#pragma unroll
  for (int e = 0; e < 16; ++e) a[e] = p[(e & 7) + ((e >> 3) << 4)];
  return a;
}

// B fragment: 32x16 bf16 tile, row-major source (K rows, N cols): N = lane&15, K same order as A.
__device__ __forceinline__ v16bf load_b(const __bf16* B, int ldb, int k0, int n0, int lane) {
  v16bf b;
  const __bf16* p = B + (k0 + ((lane >> 4) << 3)) * ldb + n0 + (lane & 15);
#pragma unroll
  for (int e = 0; e < 16; ++e) b[e] = p[((e & 7) + ((e >> 3) << 4)) * ldb];
  return b;
}

// C/D: vgpr r -> M = m0 + r + 8*(lane>>4); N = n0 + (lane&15)
__device__ __forceinline__ void store_c(float* C, int ldc, int m0, int n0, int lane, v8f c) {
  int n  = n0 + (lane & 15);
  int mh = m0 + ((lane >> 4) << 3);
#pragma unroll
  for (int r = 0; r < 8; ++r) C[(mh + r) * ldc + n] = c[r];
}

// ---------------------------------------------------------------------------
// Kernel 1: zero group counters
// ---------------------------------------------------------------------------
__global__ void zero_cnt(int* cnt) {
  int i = blockIdx.x * blockDim.x + threadIdx.x;
  if (i < NB) cnt[i] = 0;
}

// ---------------------------------------------------------------------------
// Kernel 2: bin edges into per-group slots (ts + edge id), drop overflow >64
// ---------------------------------------------------------------------------
__global__ void bin_edges(const int* __restrict__ index, const float* __restrict__ ts,
                          int* cnt, int* id_slot, float* ts_slot) {
  int e = blockIdx.x * blockDim.x + threadIdx.x;
  if (e >= NE) return;
  int g = index[e];
  int p = atomicAdd(&cnt[g], 1);
  if (p < NL) {
    id_slot[g * NL + p] = e;
    ts_slot[g * NL + p] = ts[e];
  }
}

// ---------------------------------------------------------------------------
// Kernel 3: transpose + zero-pad weights into bf16 WMMA "B" layout (K x N)
// ---------------------------------------------------------------------------
__global__ void prep_weights(const float* __restrict__ wqkv, const float* __restrict__ wout,
                             const float* __restrict__ wff1, const float* __restrict__ wff2,
                             __bf16* Wq, __bf16* Wo, __bf16* W1, __bf16* W2) {
  int i = blockIdx.x * blockDim.x + threadIdx.x;
  if (i < KPAD * QLD) { int k = i / QLD, n = i % QLD;
    Wq[i] = (k < ND && n < 3 * ND) ? f2bf(wqkv[n * ND + k]) : f2bf(0.f); }
  if (i < KPAD * ON)  { int k = i / ON,  n = i % ON;
    Wo[i] = (k < ND && n < ND)     ? f2bf(wout[n * ND + k]) : f2bf(0.f); }
  if (i < KPAD * F1N) { int k = i / F1N, n = i % F1N;
    W1[i] = (k < ND && n < 4 * ND) ? f2bf(wff1[n * ND + k]) : f2bf(0.f); }
  if (i < F1N * ON)   { int k = i / ON,  n = i % ON;
    W2[i] = (k < 4 * ND && n < ND) ? f2bf(wff2[n * 4 * ND + k]) : f2bf(0.f); }
}

// ---------------------------------------------------------------------------
// Kernel 4: one 128-thread block per group -> full fused encoder layer
// ---------------------------------------------------------------------------
__global__ __launch_bounds__(128) void tx_main(
    const float* __restrict__ x,
    const int* __restrict__ cnt, const int* __restrict__ id_slot,
    const float* __restrict__ ts_slot,
    const __bf16* __restrict__ Wq, const __bf16* __restrict__ Wo,
    const __bf16* __restrict__ W1, const __bf16* __restrict__ W2,
    const float* __restrict__ bqkv, const float* __restrict__ bout,
    const float* __restrict__ bff1, const float* __restrict__ bff2,
    const float* __restrict__ ln1g, const float* __restrict__ ln1b,
    const float* __restrict__ ln2g, const float* __restrict__ ln2b,
    float* __restrict__ out) {
  __shared__ float  hs[NL * 68];      // running hidden state (f32)
  __shared__ float  hs2[NL * ON];     // attn concat output / FF2 accumulator
  __shared__ float  cbuf[NL * 96];    // f32 GEMM output (scores / out-proj)
  __shared__ __bf16 abuf[NL * 96];    // bf16 A operand staging
  __shared__ __bf16 abuf2[NL * 96];   // attn probs / relu(ff1 chunk)
  __shared__ __bf16 qkvbf[NL * QLD];  // q|k|v bf16
  __shared__ float  skey[NL];
  __shared__ int    sid[NL];
  __shared__ int    svalid[NL];

  const int g    = blockIdx.x;
  const int tid  = threadIdx.x;
  const int lane = tid & 31;
  const int wave = tid >> 5;

  // ---- load slots; bitonic sort ascending by ts, invalid (-1) sorts first ----
  if (tid < NL) {
    int c = cnt[g]; if (c > NL) c = NL;
    if (tid < c) { skey[tid] = ts_slot[g * NL + tid]; sid[tid] = id_slot[g * NL + tid]; }
    else         { skey[tid] = -1.0f;                 sid[tid] = -1; }
  }
  __syncthreads();
  for (int k = 2; k <= NL; k <<= 1) {
    for (int j = k >> 1; j > 0; j >>= 1) {
      if (tid < NL) {
        int ixj = tid ^ j;
        if (ixj > tid) {
          bool up = ((tid & k) == 0);
          float ka = skey[tid], kb = skey[ixj];
          if ((ka > kb) == up) {
            skey[tid] = kb; skey[ixj] = ka;
            int t = sid[tid]; sid[tid] = sid[ixj]; sid[ixj] = t;
          }
        }
      }
      __syncthreads();
    }
  }
  if (tid < NL) svalid[tid] = (sid[tid] >= 0) ? 1 : 0;
  __syncthreads();

  // ---- gather x + sinusoidal positional encoding -> hs ----
  for (int i = tid; i < NL * ND; i += 128) {
    int r = i / ND, c = i % ND;
    int id = sid[r];
    float v   = (id >= 0) ? x[(long)id * ND + c] : 0.0f;
    float ang = (float)r * __expf((float)(c & ~1) * (-9.2103403719761836f / 66.0f));
    float pe  = (c & 1) ? __cosf(ang) : __sinf(ang);
    hs[r * 68 + c] = v + pe;
  }
  __syncthreads();
  for (int i = tid; i < NL * 96; i += 128) {
    int r = i / 96, c = i % 96;
    abuf[i] = (c < ND) ? f2bf(hs[r * 68 + c]) : f2bf(0.f);
  }
  __syncthreads();

  // ---- QKV projection: [64,96] x [96,208] -> qkvbf (bias fused, bf16 out) ----
  {
    int m0 = wave * 16;
    for (int nt = 0; nt < 13; ++nt) {
      v8f acc = {};
      for (int ks = 0; ks < 3; ++ks) {
        v16bf a = load_a(abuf, 96, m0, ks * 32, lane);
        v16bf b = load_b(Wq, QLD, ks * 32, nt * 16, lane);
        acc = WMMA_BF16(a, b, acc);
      }
      int n  = nt * 16 + (lane & 15);
      int mh = m0 + ((lane >> 4) << 3);
      float bv = (n < 3 * ND) ? bqkv[n] : 0.f;
#pragma unroll
      for (int r = 0; r < 8; ++r)
        qkvbf[(mh + r) * QLD + n] = (n < 3 * ND) ? f2bf(acc[r] + bv) : f2bf(0.f);
    }
  }
  __syncthreads();

  // ---- attention per head (HD=33 zero-padded to K=64) ----
  for (int h = 0; h < 2; ++h) {
    const int qoff = h * 33, koff = ND + h * 33, voff = 2 * ND + h * 33;
    // scores = q @ k^T
    for (int nt = 0; nt < 4; ++nt) {
      v8f acc = {};
      for (int ks = 0; ks < 2; ++ks) {
        v16bf a, b;
        int m  = wave * 16 + (lane & 15);
        int kb = ks * 32 + ((lane >> 4) << 3);
#pragma unroll
        for (int e = 0; e < 16; ++e) {
          int hd = kb + (e & 7) + ((e >> 3) << 4);
          a[e] = (hd < 33) ? qkvbf[m * QLD + qoff + hd] : f2bf(0.f);
        }
        int kr = nt * 16 + (lane & 15);
#pragma unroll
        for (int e = 0; e < 16; ++e) {
          int kk = kb + (e & 7) + ((e >> 3) << 4);
          b[e] = (kk < 33) ? qkvbf[kr * QLD + koff + kk] : f2bf(0.f);
        }
        acc = WMMA_BF16(a, b, acc);
      }
      store_c(cbuf, 96, wave * 16, nt * 16, lane, acc);
    }
    __syncthreads();
    // masked softmax (one row per thread)
    if (tid < NL) {
      const float scale = 0.17407765595569785f;  // 1/sqrt(33)
      float mx = -3.0e38f;
      for (int j = 0; j < NL; ++j) {
        float s = svalid[j] ? cbuf[tid * 96 + j] * scale : -1.0e9f;
        cbuf[tid * 96 + j] = s;
        mx = fmaxf(mx, s);
      }
      float sum = 0.f;
      for (int j = 0; j < NL; ++j) {
        float e2 = __expf(cbuf[tid * 96 + j] - mx);
        cbuf[tid * 96 + j] = e2;
        sum += e2;
      }
      float inv = 1.0f / sum;
      for (int j = 0; j < NL; ++j) abuf2[tid * 96 + j] = f2bf(cbuf[tid * 96 + j] * inv);
    }
    __syncthreads();
    // o = attn @ v -> hs2[:, h*33 : h*33+33]
    for (int nt = 0; nt < 3; ++nt) {
      v8f acc = {};
      for (int ks = 0; ks < 2; ++ks) {
        v16bf a = load_a(abuf2, 96, wave * 16, ks * 32, lane);
        v16bf b;
        int hd = nt * 16 + (lane & 15);
#pragma unroll
        for (int e = 0; e < 16; ++e) {
          int kr = ks * 32 + ((lane >> 4) << 3) + (e & 7) + ((e >> 3) << 4);
          b[e] = (hd < 33) ? qkvbf[kr * QLD + voff + hd] : f2bf(0.f);
        }
        acc = WMMA_BF16(a, b, acc);
      }
      int n  = nt * 16 + (lane & 15);
      int mh = wave * 16 + ((lane >> 4) << 3);
      if (n < 33) {
#pragma unroll
        for (int r = 0; r < 8; ++r) hs2[(mh + r) * ON + h * 33 + n] = acc[r];
      }
    }
    __syncthreads();
  }

  // ---- out projection + residual + LN1 ----
  for (int i = tid; i < NL * 96; i += 128) {
    int r = i / 96, c = i % 96;
    abuf[i] = (c < ND) ? f2bf(hs2[r * ON + c]) : f2bf(0.f);
  }
  __syncthreads();
  {
    int m0 = wave * 16;
    for (int nt = 0; nt < 5; ++nt) {
      v8f acc = {};
      for (int ks = 0; ks < 3; ++ks) {
        v16bf a = load_a(abuf, 96, m0, ks * 32, lane);
        v16bf b = load_b(Wo, ON, ks * 32, nt * 16, lane);
        acc = WMMA_BF16(a, b, acc);
      }
      store_c(cbuf, 96, m0, nt * 16, lane, acc);
    }
  }
  __syncthreads();
  if (tid < NL) {
    float m = 0.f;
    for (int c = 0; c < ND; ++c) m += hs[tid * 68 + c] + cbuf[tid * 96 + c] + bout[c];
    m *= (1.0f / 66.0f);
    float var = 0.f;
    for (int c = 0; c < ND; ++c) {
      float d = hs[tid * 68 + c] + cbuf[tid * 96 + c] + bout[c] - m;
      var += d * d;
    }
    var *= (1.0f / 66.0f);
    float inv = rsqrtf(var + 1e-5f);
    for (int c = 0; c < ND; ++c) {
      float d = hs[tid * 68 + c] + cbuf[tid * 96 + c] + bout[c] - m;
      hs[tid * 68 + c] = d * inv * ln1g[c] + ln1b[c];
    }
  }
  __syncthreads();

  // ---- feed-forward in 3 chunks of 96 ff1-columns; FF2 accumulated in hs2 ----
  for (int i = tid; i < NL * 96; i += 128) {
    int r = i / 96, c = i % 96;
    abuf[i] = (c < ND) ? f2bf(hs[r * 68 + c]) : f2bf(0.f);
  }
  for (int i = tid; i < NL * ON; i += 128) hs2[i] = 0.f;
  __syncthreads();
  for (int cc = 0; cc < 3; ++cc) {
    {  // ff1 chunk: relu(h @ W1[:, cc*96 : cc*96+96] + b1) -> abuf2 (bf16)
      int m0 = wave * 16;
      for (int nt = 0; nt < 6; ++nt) {
        v8f acc = {};
        for (int ks = 0; ks < 3; ++ks) {
          v16bf a = load_a(abuf, 96, m0, ks * 32, lane);
          v16bf b = load_b(W1, F1N, ks * 32, cc * 96 + nt * 16, lane);
          acc = WMMA_BF16(a, b, acc);
        }
        int nl = nt * 16 + (lane & 15);
        int ng = cc * 96 + nl;
        int mh = m0 + ((lane >> 4) << 3);
        float bv = (ng < 4 * ND) ? bff1[ng] : 0.f;
#pragma unroll
        for (int r = 0; r < 8; ++r) {
          float v = fmaxf(acc[r] + bv, 0.f);
          abuf2[(mh + r) * 96 + nl] = (ng < 4 * ND) ? f2bf(v) : f2bf(0.f);
        }
      }
    }
    __syncthreads();
    {  // ff2 partial: hs2 += abuf2 @ W2[cc*96 : cc*96+96, :]
      int m0 = wave * 16;
      for (int nt = 0; nt < 5; ++nt) {
        v8f acc = {};
        for (int ks = 0; ks < 3; ++ks) {
          v16bf a = load_a(abuf2, 96, m0, ks * 32, lane);
          v16bf b = load_b(W2, ON, cc * 96 + ks * 32, nt * 16, lane);
          acc = WMMA_BF16(a, b, acc);
        }
        int n  = nt * 16 + (lane & 15);
        int mh = m0 + ((lane >> 4) << 3);
#pragma unroll
        for (int r = 0; r < 8; ++r) hs2[(mh + r) * ON + n] += acc[r];
      }
    }
    __syncthreads();
  }

  // ---- residual + LN2 ----
  if (tid < NL) {
    float m = 0.f;
    for (int c = 0; c < ND; ++c) m += hs[tid * 68 + c] + hs2[tid * ON + c] + bff2[c];
    m *= (1.0f / 66.0f);
    float var = 0.f;
    for (int c = 0; c < ND; ++c) {
      float d = hs[tid * 68 + c] + hs2[tid * ON + c] + bff2[c] - m;
      var += d * d;
    }
    var *= (1.0f / 66.0f);
    float inv = rsqrtf(var + 1e-5f);
    for (int c = 0; c < ND; ++c) {
      float d = hs[tid * 68 + c] + hs2[tid * ON + c] + bff2[c] - m;
      hs[tid * 68 + c] = d * inv * ln2g[c] + ln2b[c];
    }
  }
  __syncthreads();

  // ---- zero padded rows, mean over L=64 ----
  if (tid < ND) {
    float s = 0.f;
    for (int r = 0; r < NL; ++r)
      if (svalid[r]) s += hs[r * 68 + tid];
    out[(long)g * ND + tid] = s * (1.0f / 64.0f);
  }
}

// ---------------------------------------------------------------------------
// Host launcher
// ---------------------------------------------------------------------------
extern "C" void kernel_launch(void* const* d_in, const int* in_sizes, int n_in,
                              void* d_out, int out_size, void* d_ws, size_t ws_size,
                              hipStream_t stream) {
  const float* x     = (const float*)d_in[0];
  const int*   index = (const int*)d_in[1];
  const float* ts    = (const float*)d_in[2];
  const float* w_qkv = (const float*)d_in[3];
  const float* b_qkv = (const float*)d_in[4];
  const float* w_out = (const float*)d_in[5];
  const float* b_out = (const float*)d_in[6];
  const float* w_ff1 = (const float*)d_in[7];
  const float* b_ff1 = (const float*)d_in[8];
  const float* w_ff2 = (const float*)d_in[9];
  const float* b_ff2 = (const float*)d_in[10];
  const float* ln1g  = (const float*)d_in[11];
  const float* ln1b  = (const float*)d_in[12];
  const float* ln2g  = (const float*)d_in[13];
  const float* ln2b  = (const float*)d_in[14];
  float* out = (float*)d_out;

  char* ws = (char*)d_ws;
  int*    cnt     = (int*)ws;    ws += (size_t)NB * 4;
  int*    id_slot = (int*)ws;    ws += (size_t)NB * NL * 4;
  float*  ts_slot = (float*)ws;  ws += (size_t)NB * NL * 4;
  __bf16* Wq      = (__bf16*)ws; ws += (size_t)KPAD * QLD * 2;
  __bf16* Wo      = (__bf16*)ws; ws += (size_t)KPAD * ON * 2;
  __bf16* W1      = (__bf16*)ws; ws += (size_t)KPAD * F1N * 2;
  __bf16* W2      = (__bf16*)ws; ws += (size_t)F1N * ON * 2;

  zero_cnt<<<(NB + 255) / 256, 256, 0, stream>>>(cnt);
  bin_edges<<<(NE + 255) / 256, 256, 0, stream>>>(index, ts, cnt, id_slot, ts_slot);
  prep_weights<<<(KPAD * F1N + 255) / 256, 256, 0, stream>>>(w_qkv, w_out, w_ff1, w_ff2,
                                                             Wq, Wo, W1, W2);
  tx_main<<<NB, 128, 0, stream>>>(x, cnt, id_slot, ts_slot, Wq, Wo, W1, W2,
                                  b_qkv, b_out, b_ff1, b_ff2,
                                  ln1g, ln1b, ln2g, ln2b, out);
}
